// MultiHeadAttention_74483322847931
// MI455X (gfx1250) — compile-verified, matrix operand order checked by reading, not run on hardware
//
#include <hip/hip_runtime.h>

typedef __attribute__((ext_vector_type(16))) _Float16 v16h;
typedef __attribute__((ext_vector_type(8)))  _Float16 v8h;
typedef __attribute__((ext_vector_type(8)))  float    v8f;
typedef __attribute__((ext_vector_type(4)))  unsigned int u32x4;
typedef __attribute__((ext_vector_type(8)))  unsigned int u32x8;

#define D_MODEL   1024
#define NUM_HEADS 16
#define D_K       64
#define SEQ       2048
#define BATCH     4

// ---------------------------------------------------------------------------
// Tensor Data Mover: async 2D f16 tile load Global -> LDS (CDNA5 ISA).
// D# group0: count=1 | lds_addr | global_addr[56:0] | type=2.
// D# group1: data_size=2B, pad_enable, pad_interval (2^(v+1) dwords),
//            pad_amount (v+1 dwords), tensor dims/strides, tile dims.
// Issued once per wave (EXEC-independent); caller waits on TENSORcnt.
// ---------------------------------------------------------------------------
__device__ __forceinline__ void tdm_load_2d(unsigned lds_off, const void* gaddr,
                                            unsigned tensor_d0, unsigned tensor_d1,
                                            unsigned tile_d0, unsigned tile_d1,
                                            unsigned stride_d0,
                                            unsigned pad_interval, unsigned pad_amount) {
  unsigned long long ga = (unsigned long long)gaddr;
  u32x4 g0;
  g0[0] = 1u;                                                  // count=1 (user D#)
  g0[1] = lds_off;                                             // LDS byte address
  g0[2] = (unsigned)ga;                                        // global_addr[31:0]
  g0[3] = (unsigned)((ga >> 32) & 0x01FFFFFFu) | (2u << 30);   // [56:32] | type=2
  u32x8 g1;
  g1[0] = (1u << 16)                                           // data_size = 2 bytes
        | (1u << 20)                                           // pad_enable
        | ((pad_interval & 7u) << 22)
        | ((pad_amount & 127u) << 25);
  g1[1] = (tensor_d0 & 0xFFFFu) << 16;                         // tensor_dim0[15:0] -> [63:48]
  g1[2] = ((tensor_d0 >> 16) & 0xFFFFu)                        // tensor_dim0[31:16] -> [79:64]
        | ((tensor_d1 & 0xFFFFu) << 16);                       // tensor_dim1[15:0] -> [95:80]
  g1[3] = ((tensor_d1 >> 16) & 0xFFFFu)                        // tensor_dim1[31:16] -> [111:96]
        | ((tile_d0 & 0xFFFFu) << 16);                         // tile_dim0 -> [127:112]
  g1[4] = (tile_d1 & 0xFFFFu);                                 // tile_dim1; tile_dim2 = 0
  g1[5] = stride_d0;                                           // tensor_dim0_stride[31:0]
  g1[6] = 0u;
  g1[7] = 0u;
  asm volatile("tensor_load_to_lds %0, %1, null, null" :: "s"(g0), "s"(g1) : "memory");
}

// K index inside a 16x32 (f16) WMMA fragment per CDNA5 ISA layout.
__device__ __forceinline__ int frag_k(int v, int hs) {
  return ((v & 4) ? 16 : 0) + hs * 8 + (v & 3) * 2;
}

// A fragment (16x32 f16, M = lane&15) from LDS row-major tile.
__device__ __forceinline__ v16h frag_a_lds(const _Float16* base, int stride, int k_off, int lane) {
  v16h f;
  const int hs = (lane >> 4) & 1;
  const _Float16* row = base + (lane & 15) * stride + k_off;
#pragma unroll
  for (int v = 0; v < 8; ++v) {
    int k = frag_k(v, hs);
    f[2 * v]     = row[k];
    f[2 * v + 1] = row[k + 1];
  }
  return f;
}

// B fragment (32x16 f16, N = lane&15) from LDS row-major [k][n] tile.
__device__ __forceinline__ v16h frag_b_lds(const _Float16* base, int stride, int n_off, int lane) {
  v16h f;
  const int hs = (lane >> 4) & 1;
  const int n  = (lane & 15) + n_off;
#pragma unroll
  for (int v = 0; v < 8; ++v) {
    int k = frag_k(v, hs);
    f[2 * v]     = base[k * stride + n];
    f[2 * v + 1] = base[(k + 1) * stride + n];
  }
  return f;
}

// Elementwise f32 -> f16 (x and weights converted exactly once).
__global__ __launch_bounds__(256) void cvt_f16(const float* __restrict__ src,
                                               _Float16* __restrict__ dst, size_t n) {
  size_t i = (size_t)blockIdx.x * blockDim.x + threadIdx.x;
  size_t stride = (size_t)gridDim.x * blockDim.x;
  for (; i < n; i += stride) dst[i] = (_Float16)src[i];
}

// ---------------------------------------------------------------------------
// GEMM: C[M,N] = A[M,K] (f16) x W[K,N] (f16) + bias (f32).
// Block tile 64x128, 8 waves as 2(M) x 4(N); each wave: 32x32 = 4 WMMAs/k-step
// with 2x2 fragment reuse. Tiles arrive via TDM (wave 0 issues, TENSORcnt).
// ---------------------------------------------------------------------------
template <bool OUT_HALF>
__global__ __launch_bounds__(256) void gemm_bias(const _Float16* __restrict__ A,
                                                 const _Float16* __restrict__ W,
                                                 const float* __restrict__ bias,
                                                 void* __restrict__ Cout,
                                                 int M, int N, int K) {
  __shared__ _Float16 sA[64 * 40];   // 64x32, TDM-padded stride 40
  __shared__ _Float16 sW[32 * 136];  // 32x128, TDM-padded stride 136

  const int t    = threadIdx.x;
  const int lane = t & 31;
  const int wv   = t >> 5;
  const int m0   = blockIdx.y * 64;
  const int n0   = blockIdx.x * 128;
  const int m_sub = wv & 1;
  const int n_sub = wv >> 1;

  const unsigned offA = (unsigned)(uintptr_t)&sA[0];
  const unsigned offW = (unsigned)(uintptr_t)&sW[0];

  v8f acc00 = {}, acc01 = {}, acc10 = {}, acc11 = {};

  for (int k0 = 0; k0 < K; k0 += 32) {
    if (t < 32) {
      // A tile: 32 elems/row (16 dw -> interval code 3), pad 4 dw (code 3) -> stride 40
      tdm_load_2d(offA, A + (size_t)m0 * K + k0, K, M, 32, 64, K, 3, 3);
      // W tile: 128 elems/row (64 dw -> interval code 5), pad 4 dw (code 3) -> stride 136
      tdm_load_2d(offW, W + (size_t)k0 * N + n0, N, K, 128, 32, N, 5, 3);
      __builtin_amdgcn_s_wait_tensorcnt(0);
    }
    __syncthreads();

    v16h a0 = frag_a_lds(sA + (m_sub * 32) * 40, 40, 0, lane);
    v16h a1 = frag_a_lds(sA + (m_sub * 32 + 16) * 40, 40, 0, lane);
    v16h b0 = frag_b_lds(sW, 136, n_sub * 32, lane);
    v16h b1 = frag_b_lds(sW, 136, n_sub * 32 + 16, lane);
    acc00 = __builtin_amdgcn_wmma_f32_16x16x32_f16(false, a0, false, b0, (short)0, acc00, false, false);
    acc01 = __builtin_amdgcn_wmma_f32_16x16x32_f16(false, a0, false, b1, (short)0, acc01, false, false);
    acc10 = __builtin_amdgcn_wmma_f32_16x16x32_f16(false, a1, false, b0, (short)0, acc10, false, false);
    acc11 = __builtin_amdgcn_wmma_f32_16x16x32_f16(false, a1, false, b1, (short)0, acc11, false, false);
    __syncthreads();
  }

  const int hs = (lane >> 4) & 1;
  const int n  = n0 + n_sub * 32 + (lane & 15);
  const float bv0 = bias[n];
  const float bv1 = bias[n + 16];
#pragma unroll
  for (int v = 0; v < 8; ++v) {
    int m = m0 + m_sub * 32 + hs * 8 + v;
    float r00 = acc00[v] + bv0, r01 = acc01[v] + bv1;
    float r10 = acc10[v] + bv0, r11 = acc11[v] + bv1;
    if (OUT_HALF) {
      _Float16* C = (_Float16*)Cout;
      C[(size_t)m * N + n]             = (_Float16)r00;
      C[(size_t)m * N + n + 16]        = (_Float16)r01;
      C[(size_t)(m + 16) * N + n]      = (_Float16)r10;
      C[(size_t)(m + 16) * N + n + 16] = (_Float16)r11;
    } else {
      float* C = (float*)Cout;
      C[(size_t)m * N + n]             = r00;
      C[(size_t)m * N + n + 16]        = r01;
      C[(size_t)(m + 16) * N + n]      = r10;
      C[(size_t)(m + 16) * N + n + 16] = r11;
    }
  }
}

// ---------------------------------------------------------------------------
// Attention: one block = (b, h, 16-row q-tile).
// Q tile via TDM; scores(16x2048) by WMMA into 128KB LDS; block softmax;
// probs streamed coalesced (f32) + cached f16 in LDS; PV by WMMA with
// per-wave b128-staged V tiles; partials merged with ds_add_f32.
// ---------------------------------------------------------------------------
__global__ __launch_bounds__(256) void attention_kernel(const _Float16* __restrict__ Qh,
                                                        const _Float16* __restrict__ Kh,
                                                        const _Float16* __restrict__ Vh,
                                                        const int* __restrict__ mask,
                                                        float* __restrict__ probs,
                                                        _Float16* __restrict__ attn_out) {
  __shared__ float    s_scores[16][SEQ];       // 128 KB
  __shared__ _Float16 s_p16[16][2056];         // 64 KB: probs as f16 for PV
  __shared__ _Float16 s_q[16][72];             // TDM-padded stride 72
  __shared__ _Float16 s_vstage[8][32][24];     // per-wave V staging (16B-aligned rows)
  __shared__ float    s_red[16][17];
  __shared__ float    s_rowmax[16];
  __shared__ float    s_rowsum[16];
  __shared__ float    s_pv[16][64];

  const int t    = threadIdx.x;
  const int lane = t & 31;
  const int wv   = t >> 5;
  const int tiles_q = SEQ / 16;  // 128

  const int qt = blockIdx.x % tiles_q;
  const int h  = (blockIdx.x / tiles_q) % NUM_HEADS;
  const int b  = blockIdx.x / (tiles_q * NUM_HEADS);
  const int q0 = qt * 16;

  if (t < 32) {
    // Q tile 64 elems/row (32 dw -> interval code 4), pad 4 dw (code 3) -> stride 72
    tdm_load_2d((unsigned)(uintptr_t)&s_q[0][0],
                Qh + (size_t)(b * SEQ + q0) * D_MODEL + h * D_K,
                D_MODEL, BATCH * SEQ, D_K, 16, D_MODEL, 4, 3);
    __builtin_amdgcn_s_wait_tensorcnt(0);
  }
  __syncthreads();

  const int hs = (lane >> 4) & 1;
  const int nn = lane & 15;
  const float scale = 0.125f;  // 1/sqrt(64)

  // ---- Phase 1: scores = Q K^T * scale (+mask) ----
  v16h qa0 = frag_a_lds(&s_q[0][0], 72, 0, lane);
  v16h qa1 = frag_a_lds(&s_q[0][0], 72, 32, lane);
  for (int kt = 0; kt < 16; ++kt) {
    const int key0 = (wv * 16 + kt) * 16;  // 8 waves x 16 tiles cover 2048 keys
    const _Float16* kb = Kh + (size_t)(b * SEQ + key0 + nn) * D_MODEL + h * D_K;
    __builtin_prefetch(kb + 16 * D_MODEL, 0, 1);  // next key tile -> global_prefetch
    v16h kf0, kf1;
#pragma unroll
    for (int v = 0; v < 8; ++v) {
      int k = frag_k(v, hs);
      kf0[2 * v]     = kb[k];
      kf0[2 * v + 1] = kb[k + 1];
      kf1[2 * v]     = kb[32 + k];
      kf1[2 * v + 1] = kb[32 + k + 1];
    }
    v8f acc = {};
    acc = __builtin_amdgcn_wmma_f32_16x16x32_f16(false, qa0, false, kf0, (short)0, acc, false, false);
    acc = __builtin_amdgcn_wmma_f32_16x16x32_f16(false, qa1, false, kf1, (short)0, acc, false, false);
#pragma unroll
    for (int v = 0; v < 8; ++v) {
      int m   = hs * 8 + v;
      int col = key0 + nn;
      int mk  = mask[((size_t)b * SEQ + q0 + m) * SEQ + col];
      float sc = acc[v] * scale;
      s_scores[m][col] = (mk == 0) ? -1.0e9f : sc;
    }
  }
  __syncthreads();

  // ---- Phase 2: softmax (16 threads per row) ----
  {
    int r = t >> 4, sub = t & 15;
    float mx = -3.0e38f;
    for (int c = sub * 128; c < sub * 128 + 128; ++c) mx = fmaxf(mx, s_scores[r][c]);
    s_red[r][sub] = mx;
  }
  __syncthreads();
  if (t < 16) {
    float mx = s_red[t][0];
#pragma unroll
    for (int i = 1; i < 16; ++i) mx = fmaxf(mx, s_red[t][i]);
    s_rowmax[t] = mx;
  }
  __syncthreads();
  {
    int r = t >> 4, sub = t & 15;
    float rm = s_rowmax[r], sm = 0.f;
    for (int c = sub * 128; c < sub * 128 + 128; ++c) sm += __expf(s_scores[r][c] - rm);
    s_red[r][sub] = sm;
  }
  __syncthreads();
  if (t < 16) {
    float sm = 0.f;
#pragma unroll
    for (int i = 0; i < 16; ++i) sm += s_red[t][i];
    s_rowsum[t] = sm;
  }
  __syncthreads();
  {  // normalize: stream f32 probs coalesced + cache f16 copy for PV
    float* prow = probs + ((size_t)(b * NUM_HEADS + h) * SEQ + q0) * SEQ;
#pragma unroll 4
    for (int j = 0; j < 128; ++j) {
      int idx = t + 256 * j;
      int rr = idx >> 11, cc = idx & 2047;
      float p = __expf(s_scores[rr][cc] - s_rowmax[rr]) / s_rowsum[rr];
      s_p16[rr][cc] = (_Float16)p;
      prow[(size_t)rr * SEQ + cc] = p;
    }
  }
  __syncthreads();

  // ---- Phase 3: out = P @ V ----
#pragma unroll
  for (int j = 0; j < 4; ++j) ((float*)s_pv)[t + 256 * j] = 0.f;
  __syncthreads();
  {
    const int ntile = wv & 3;   // d_k column tile of 16
    const int khalf = wv >> 2;  // 2 waves split the 2048-key reduction
    const int nb = ntile * 16;
    _Float16* stage = &s_vstage[wv][0][0];
    v8f acc = {};
    for (int chunk = 0; chunk < 32; ++chunk) {
      const int kb0 = khalf * 1024 + chunk * 32;
      // stage this wave's 32x16 V sub-tile: two b128 loads + b128 LDS stores
      const v8h* vsrc = (const v8h*)(Vh + (size_t)(b * SEQ + kb0 + lane) * D_MODEL + h * D_K + nb);
      v8h* vdst = (v8h*)(stage + lane * 24);
      vdst[0] = vsrc[0];
      vdst[1] = vsrc[1];
      v16h pa = frag_a_lds(&s_p16[0][0], 2056, kb0, lane);
      v16h vb = frag_b_lds(stage, 24, 0, lane);
      acc = __builtin_amdgcn_wmma_f32_16x16x32_f16(false, pa, false, vb, (short)0, acc, false, false);
    }
#pragma unroll
    for (int v = 0; v < 8; ++v) atomicAdd(&s_pv[hs * 8 + v][nb + nn], acc[v]);  // ds_add_f32
  }
  __syncthreads();
#pragma unroll
  for (int j = 0; j < 4; ++j) {
    int idx = t + 256 * j;
    int r = idx >> 6, c = idx & 63;
    attn_out[(size_t)(b * SEQ + q0 + r) * D_MODEL + h * D_K + c] = (_Float16)s_pv[r][c];
  }
}

extern "C" void kernel_launch(void* const* d_in, const int* in_sizes, int n_in,
                              void* d_out, int out_size, void* d_ws, size_t ws_size,
                              hipStream_t stream) {
  const float* x  = (const float*)d_in[0];
  const int* mask = (const int*)d_in[1];
  const float* Wq = (const float*)d_in[2];
  const float* bq = (const float*)d_in[3];
  const float* Wk = (const float*)d_in[4];
  const float* bk = (const float*)d_in[5];
  const float* Wv = (const float*)d_in[6];
  const float* bv = (const float*)d_in[7];
  const float* Wo = (const float*)d_in[8];
  const float* bo = (const float*)d_in[9];

  const int M = BATCH * SEQ;  // 8192
  const int N = D_MODEL;      // 1024
  const int K = D_MODEL;      // 1024

  // Workspace layout (f16 halves): xh | wq..wo | Q | K | V | attn  (~92 MB)
  _Float16* xh  = (_Float16*)d_ws;
  _Float16* wqh = xh  + (size_t)M * K;
  _Float16* wkh = wqh + (size_t)K * N;
  _Float16* wvh = wkh + (size_t)K * N;
  _Float16* woh = wvh + (size_t)K * N;
  _Float16* Qh  = woh + (size_t)K * N;
  _Float16* Kh  = Qh  + (size_t)M * N;
  _Float16* Vh  = Kh  + (size_t)M * N;
  _Float16* attn = Vh + (size_t)M * N;

  float* out   = (float*)d_out;
  float* probs = out + (size_t)M * D_MODEL;  // tuple: (out, probs)

  cvt_f16<<<2048, 256, 0, stream>>>(x,  xh,  (size_t)M * K);
  cvt_f16<<<512,  256, 0, stream>>>(Wq, wqh, (size_t)K * N);
  cvt_f16<<<512,  256, 0, stream>>>(Wk, wkh, (size_t)K * N);
  cvt_f16<<<512,  256, 0, stream>>>(Wv, wvh, (size_t)K * N);
  cvt_f16<<<512,  256, 0, stream>>>(Wo, woh, (size_t)K * N);

  dim3 gblk(N / 128, M / 64);
  gemm_bias<true><<<gblk, 256, 0, stream>>>(xh, wqh, bq, Qh, M, N, K);
  gemm_bias<true><<<gblk, 256, 0, stream>>>(xh, wkh, bk, Kh, M, N, K);
  gemm_bias<true><<<gblk, 256, 0, stream>>>(xh, wvh, bv, Vh, M, N, K);

  attention_kernel<<<BATCH * NUM_HEADS * (SEQ / 16), 256, 0, stream>>>(Qh, Kh, Vh, mask, probs, attn);

  gemm_bias<false><<<gblk, 256, 0, stream>>>(attn, woh, bo, out, M, N, K);
}
